// Mamba_63720134803675
// MI455X (gfx1250) — matrix operand order repaired and checked by reading, not compiled, not run on hardware
//
#include <hip/hip_runtime.h>
#include <math.h>

// ---------------------------------------------------------------------------
// Mamba forward for MI455X (gfx1250, wave32).
//   GEMMs : v_wmma_f32_16x16x32_f16, 128x64 block tile (8 waves x 32x32 each)
//   conv  : elementwise depthwise conv + SiLU
//   scan  : channel-parallel recurrence; per-timestep B/C double-buffered into
//           LDS with global_load_async_to_lds_b32 (ASYNCcnt) when available.
// ---------------------------------------------------------------------------

typedef __attribute__((ext_vector_type(16))) _Float16 v16h;
typedef __attribute__((ext_vector_type(8)))  float    v8f;

#define D_MODEL 1024
#define D_INNER 2048
#define D_STATE 16
#define D_CONV  4
#define DT_RANK 64
#define BATCH   2
#define SEQLEN  1024
#define NROWS   (BATCH * SEQLEN)          // 2048 GEMM rows
#define XPROJ_N (DT_RANK + 2 * D_STATE)   // 96

// ---- async global->LDS copy (gfx1250), with compile-safe fallback ---------
// Probe-confirmed this round: builtin exists and takes (int AS1*, int AS3*,
// imm offset, imm cpol).
#if __has_builtin(__builtin_amdgcn_global_load_async_to_lds_b32)
#define HAVE_ASYNC_LDS 1
#define ASYNC_COPY_B32(dst_lds, src_glb)                                      \
    __builtin_amdgcn_global_load_async_to_lds_b32(                            \
        (__attribute__((address_space(1))) int*)(src_glb),                    \
        (__attribute__((address_space(3))) int*)(dst_lds), 0, 0)
#if __has_builtin(__builtin_amdgcn_s_wait_asynccnt)
#define WAIT_ASYNC() __builtin_amdgcn_s_wait_asynccnt(0)
#else
#define WAIT_ASYNC() asm volatile("s_wait_asynccnt 0x0" ::: "memory")
#endif
#else
#define HAVE_ASYNC_LDS 0
#define ASYNC_COPY_B32(dst_lds, src_glb) (*(dst_lds) = *(src_glb))
#define WAIT_ASYNC() ((void)0)
#endif

// ===========================================================================
// WMMA GEMM:  C[M,N] = act( A[M,K] (row stride lda) * W[N,K]^T + bias )
// Block tile 128x64, 256 threads = 8 waves arranged 4(M) x 2(N);
// each wave owns a 32x32 sub-tile -> four v_wmma_f32_16x16x32_f16 per K-step.
// act: 0 = identity, 1 = softplus (for dt_proj).
// ===========================================================================
__global__ __launch_bounds__(256)
void wmma_gemm_f16(const float* __restrict__ A, int lda,
                   const float* __restrict__ W,
                   const float* __restrict__ bias,
                   float* __restrict__ C,
                   int M, int N, int K, int act)
{
    __shared__ _Float16 As[128][32];  // A tile, row-major (m, k)  8 KB
    __shared__ _Float16 Ws[64][32];   // W tile, row-major (n, k)  4 KB

    const int tid  = threadIdx.x;
    const int lane = tid & 31;
    const int wid  = tid >> 5;
    const int wm   = wid & 3;     // wave M sub-tile 0..3 (32 rows each)
    const int wn   = wid >> 2;    // wave N sub-tile 0..1 (32 cols each)
    const int lm   = lane & 15;
    const int lhi  = lane >> 4;   // lane half (0: lanes 0-15, 1: lanes 16-31)

    const int row0 = blockIdx.y * 128;
    const int col0 = blockIdx.x * 64;

    v8f acc00 = {}, acc01 = {}, acc10 = {}, acc11 = {};

    for (int k0 = 0; k0 < K; k0 += 32) {
        // ---- stage A tile: 128x32 f32 -> f16, 16 contiguous elems/thread ---
        {
            const int r  = tid >> 1;            // row within tile (0..127)
            const int kk = (tid & 1) << 4;      // k within tile (0 or 16)
            const float* src = A + (size_t)(row0 + r) * lda + k0 + kk;
            const float4 f0 = *(const float4*)(src);
            const float4 f1 = *(const float4*)(src + 4);
            const float4 f2 = *(const float4*)(src + 8);
            const float4 f3 = *(const float4*)(src + 12);
            _Float16* dst = &As[r][kk];
            dst[0]  = (_Float16)f0.x; dst[1]  = (_Float16)f0.y;
            dst[2]  = (_Float16)f0.z; dst[3]  = (_Float16)f0.w;
            dst[4]  = (_Float16)f1.x; dst[5]  = (_Float16)f1.y;
            dst[6]  = (_Float16)f1.z; dst[7]  = (_Float16)f1.w;
            dst[8]  = (_Float16)f2.x; dst[9]  = (_Float16)f2.y;
            dst[10] = (_Float16)f2.z; dst[11] = (_Float16)f2.w;
            dst[12] = (_Float16)f3.x; dst[13] = (_Float16)f3.y;
            dst[14] = (_Float16)f3.z; dst[15] = (_Float16)f3.w;
        }
        // ---- stage W tile: 64x32, 8 elems/thread (guard N for x_proj) ------
        {
            const int r  = tid >> 2;            // n within tile (0..63)
            const int kk = (tid & 3) << 3;      // k within tile
            const int n  = col0 + r;
            float4 f0 = {0.f, 0.f, 0.f, 0.f};
            float4 f1 = {0.f, 0.f, 0.f, 0.f};
            if (n < N) {
                const float* src = W + (size_t)n * K + k0 + kk;
                f0 = *(const float4*)(src);
                f1 = *(const float4*)(src + 4);
            }
            _Float16* dst = &Ws[r][kk];
            dst[0] = (_Float16)f0.x; dst[1] = (_Float16)f0.y;
            dst[2] = (_Float16)f0.z; dst[3] = (_Float16)f0.w;
            dst[4] = (_Float16)f1.x; dst[5] = (_Float16)f1.y;
            dst[6] = (_Float16)f1.z; dst[7] = (_Float16)f1.w;
        }
        __syncthreads();

        // prefetch next K tile into L2 (global_prefetch_b8)
        if (k0 + 32 < K) {
            __builtin_prefetch(A + (size_t)(row0 + (tid >> 1)) * lda + k0 + 32, 0, 1);
            if (col0 + (tid >> 2) < N)
                __builtin_prefetch(W + (size_t)(col0 + (tid >> 2)) * K + k0 + 32, 0, 1);
        }

        // ---- fragments per CDNA5 wave32 WMMA layout ------------------------
        // A 16x32 f16: lane m = lane&15; half h -> K = ((h>>3)<<4)+(lhi<<3)+(h&7)
        // B 32x16 f16: lane n = lane&15; half h -> K = ((h>>3)<<4)+(lhi<<3)+(h&7)
        v16h a0, a1, b0, b1;
        #pragma unroll
        for (int h = 0; h < 16; ++h) {
            const int ka = ((h >> 3) << 4) + (lhi << 3) + (h & 7);
            a0[h] = As[wm * 32 + lm][ka];
            a1[h] = As[wm * 32 + 16 + lm][ka];
            const int kb = ((h >> 3) << 4) + (lhi << 3) + (h & 7);
            b0[h] = Ws[wn * 32 + lm][kb];
            b1[h] = Ws[wn * 32 + 16 + lm][kb];
        }
        acc00 = __builtin_amdgcn_wmma_f32_16x16x32_f16(
            false, a0, false, b0, (short)0, acc00, false, false);
        acc01 = __builtin_amdgcn_wmma_f32_16x16x32_f16(
            false, a0, false, b1, (short)0, acc01, false, false);
        acc10 = __builtin_amdgcn_wmma_f32_16x16x32_f16(
            false, a1, false, b0, (short)0, acc10, false, false);
        acc11 = __builtin_amdgcn_wmma_f32_16x16x32_f16(
            false, a1, false, b1, (short)0, acc11, false, false);
        __syncthreads();
    }

    // ---- epilogue: C/D layout -> lane n = lane&15, row m = r + 8*lhi --------
    const int mbase = row0 + wm * 32 + lhi * 8;
    const int nb0   = col0 + wn * 32 + lm;
    const int nb1   = col0 + wn * 32 + 16 + lm;
    #pragma unroll
    for (int r = 0; r < 8; ++r) {
        const int m0 = mbase + r;       // rows of a0 sub-tiles
        const int m1 = m0 + 16;         // rows of a1 sub-tiles
        float v;
        if (nb0 < N) {
            v = acc00[r];
            if (bias) v += bias[nb0];
            if (act == 1) v = (v > 20.f) ? v : log1pf(__expf(v));
            C[(size_t)m0 * N + nb0] = v;
            v = acc10[r];
            if (bias) v += bias[nb0];
            if (act == 1) v = (v > 20.f) ? v : log1pf(__expf(v));
            C[(size_t)m1 * N + nb0] = v;
        }
        if (nb1 < N) {
            v = acc01[r];
            if (bias) v += bias[nb1];
            if (act == 1) v = (v > 20.f) ? v : log1pf(__expf(v));
            C[(size_t)m0 * N + nb1] = v;
            v = acc11[r];
            if (bias) v += bias[nb1];
            if (act == 1) v = (v > 20.f) ? v : log1pf(__expf(v));
            C[(size_t)m1 * N + nb1] = v;
        }
    }
}

// ===========================================================================
// Depthwise causal conv (4 taps) + bias + SiLU.  x lives in xz[:, :D_INNER].
// ===========================================================================
__global__ __launch_bounds__(256)
void conv_silu_kernel(const float* __restrict__ xz,
                      const float* __restrict__ convw,
                      const float* __restrict__ convb,
                      float* __restrict__ xc)
{
    const int idx = blockIdx.x * 256 + threadIdx.x;   // over BATCH*SEQLEN*D_INNER
    const int d = idx & (D_INNER - 1);
    const int l = (idx >> 11) & (SEQLEN - 1);
    const int b = idx >> 21;

    float acc = convb[d];
    #pragma unroll
    for (int k = 0; k < D_CONV; ++k) {
        const int ls = l + k - (D_CONV - 1);
        if (ls >= 0)
            acc += convw[d * D_CONV + k] *
                   xz[(size_t)(b * SEQLEN + ls) * (2 * D_INNER) + d];
    }
    xc[idx] = acc / (1.f + __expf(-acc));   // SiLU
}

// ===========================================================================
// Selective scan: one thread per (batch, channel), 16-element state in regs.
// B_t/C_t (uniform per timestep) are double-buffered into LDS with async
// global->LDS copies: timestep l+1 is in flight while l is computed.
// Fuses y += u*D_skip and y *= SiLU(z).
// ===========================================================================
__global__ __launch_bounds__(256)
void selective_scan_kernel(const float* __restrict__ delta,
                           const float* __restrict__ x_dbl,
                           const float* __restrict__ xc,
                           const float* __restrict__ xz,
                           const float* __restrict__ A_log,
                           const float* __restrict__ D_skip,
                           float* __restrict__ y)
{
    __shared__ float sBC[2][2 * D_STATE];   // [buf][0..15]=B_t, [16..31]=C_t

    const int d = blockIdx.x * 256 + threadIdx.x;   // channel
    const int b = blockIdx.y;                       // batch

    float Arow[D_STATE];
    #pragma unroll
    for (int n = 0; n < D_STATE; ++n)
        Arow[n] = -__expf(A_log[d * D_STATE + n]);
    const float dskip = D_skip[d];

    float s[D_STATE];
    #pragma unroll
    for (int n = 0; n < D_STATE; ++n) s[n] = 0.f;

    // prologue: kick off async copy of timestep 0's B/C (wave 0 only)
    if (threadIdx.x < 2 * D_STATE) {
        ASYNC_COPY_B32(&sBC[0][threadIdx.x],
                       x_dbl + (size_t)(b * SEQLEN) * XPROJ_N + DT_RANK + threadIdx.x);
    }

    for (int l = 0; l < SEQLEN; ++l) {
        WAIT_ASYNC();          // wave 0: B/C for step l landed in LDS
        __syncthreads();       // make it visible to all waves
        const int cur = l & 1;

        // issue async copy for step l+1 into the other buffer
        if (l + 1 < SEQLEN && threadIdx.x < 2 * D_STATE) {
            ASYNC_COPY_B32(&sBC[cur ^ 1][threadIdx.x],
                           x_dbl + (size_t)(b * SEQLEN + l + 1) * XPROJ_N +
                               DT_RANK + threadIdx.x);
        }

        const size_t row = (size_t)(b * SEQLEN + l);
        const float dt = delta[row * D_INNER + d];
        const float u  = xc[row * D_INNER + d];
        const float du = dt * u;
        float yt = 0.f;
        #pragma unroll
        for (int n = 0; n < D_STATE; ++n) {
            const float dA = __expf(dt * Arow[n]);
            s[n] = s[n] * dA + du * sBC[cur][n];
            yt += s[n] * sBC[cur][D_STATE + n];
        }
        const float z = xz[row * (2 * D_INNER) + D_INNER + d];
        y[row * D_INNER + d] = (yt + u * dskip) * (z / (1.f + __expf(-z)));
    }
}

// ===========================================================================
// Host-side orchestration
// ===========================================================================
extern "C" void kernel_launch(void* const* d_in, const int* in_sizes, int n_in,
                              void* d_out, int out_size, void* d_ws, size_t ws_size,
                              hipStream_t stream)
{
    (void)in_sizes; (void)n_in; (void)out_size; (void)ws_size;

    const float* hidden    = (const float*)d_in[0];  // (B, L, 1024)
    const float* in_proj_w = (const float*)d_in[1];  // (4096, 1024)
    const float* conv1d_w  = (const float*)d_in[2];  // (2048, 4)
    const float* conv1d_b  = (const float*)d_in[3];  // (2048,)
    const float* x_proj_w  = (const float*)d_in[4];  // (96, 2048)
    const float* dt_proj_w = (const float*)d_in[5];  // (2048, 64)
    const float* dt_proj_b = (const float*)d_in[6];  // (2048,)
    const float* A_log     = (const float*)d_in[7];  // (2048, 16)
    const float* D_skip    = (const float*)d_in[8];  // (2048,)
    const float* out_proj_w= (const float*)d_in[9];  // (1024, 2048)
    float* out = (float*)d_out;                      // (B, L, 1024)

    // workspace carve-out (all 256B aligned)
    char* ws = (char*)d_ws;
    const size_t XZ_BYTES   = (size_t)NROWS * 2 * D_INNER * sizeof(float); // 32 MB
    const size_t XC_BYTES   = (size_t)NROWS * D_INNER * sizeof(float);     // 16 MB
    const size_t XDBL_BYTES = (((size_t)NROWS * XPROJ_N * sizeof(float)) + 255) & ~(size_t)255;
    float* xz    = (float*)(ws);
    float* xc    = (float*)(ws + XZ_BYTES);
    float* x_dbl = (float*)(ws + XZ_BYTES + XC_BYTES);
    float* delta = (float*)(ws + XZ_BYTES + XC_BYTES + XDBL_BYTES);
    float* yfin  = (float*)(ws + XZ_BYTES + XC_BYTES + XDBL_BYTES + XC_BYTES);

    // 1) xz = hidden @ in_proj_w^T        [2048 x 4096, K=1024]
    wmma_gemm_f16<<<dim3(2 * D_INNER / 64, NROWS / 128), 256, 0, stream>>>(
        hidden, D_MODEL, in_proj_w, nullptr, xz,
        NROWS, 2 * D_INNER, D_MODEL, 0);

    // 2) xc = SiLU(depthwise_conv(x) + b)
    conv_silu_kernel<<<(NROWS * D_INNER) / 256, 256, 0, stream>>>(
        xz, conv1d_w, conv1d_b, xc);

    // 3) x_dbl = xc @ x_proj_w^T          [2048 x 96, K=2048]
    wmma_gemm_f16<<<dim3((XPROJ_N + 63) / 64, NROWS / 128), 256, 0, stream>>>(
        xc, D_INNER, x_proj_w, nullptr, x_dbl,
        NROWS, XPROJ_N, D_INNER, 0);

    // 4) delta = softplus(dt_lr @ dt_proj_w^T + dt_proj_b)  [2048 x 2048, K=64]
    //    dt_lr = x_dbl[:, :64] -> same buffer with lda = 96
    wmma_gemm_f16<<<dim3(D_INNER / 64, NROWS / 128), 256, 0, stream>>>(
        x_dbl, XPROJ_N, dt_proj_w, dt_proj_b, delta,
        NROWS, D_INNER, DT_RANK, 1);

    // 5) selective scan + D-skip + SiLU(z) gating
    selective_scan_kernel<<<dim3(D_INNER / 256, BATCH), 256, 0, stream>>>(
        delta, x_dbl, xc, xz, A_log, D_skip, yfin);

    // 6) out = y @ out_proj_w^T           [2048 x 1024, K=2048]
    wmma_gemm_f16<<<dim3(D_MODEL / 64, NROWS / 128), 256, 0, stream>>>(
        yfin, D_INNER, out_proj_w, nullptr, out,
        NROWS, D_MODEL, D_INNER, 0);
}
